// TGAT_22617297781312
// MI455X (gfx1250) — matrix operand order repaired and checked by reading, not compiled, not run on hardware
//
#include <hip/hip_runtime.h>
#include <hip/hip_bf16.h>

// ---------------------------------------------------------------------------
// TGAT forward for MI455X (gfx1250).
//  - Dense projections: wave32 WMMA (v_wmma_f32_16x16x32_f16), one wave
//    computes a 16x32 f32 tile (A fragment reused across two WMMAs).
//  - Edge softmax/aggregation: two L2-resident scatter passes using
//    encoded-float atomicMax + f32 atomicAdd; normalization deferred and
//    fused into the f16 convert (associativity of softmax normalization).
// ---------------------------------------------------------------------------

typedef __attribute__((ext_vector_type(8)))  _Float16 v8h;
typedef __attribute__((ext_vector_type(16))) _Float16 v16h;
typedef __attribute__((ext_vector_type(8)))  float    v8f;

static constexpr int HID = 256;
static constexpr int TD  = 32;
static constexpr int HPD = 32;   // head dim
static constexpr int NH  = 8;    // heads

// ---------------- helpers: monotonic float<->uint encoding -----------------
__device__ __forceinline__ unsigned fenc(float f) {
    unsigned u = __float_as_uint(f);
    return (u & 0x80000000u) ? ~u : (u | 0x80000000u);
}
__device__ __forceinline__ float fdec(unsigned u) {
    unsigned b = (u & 0x80000000u) ? (u & 0x7FFFFFFFu) : ~u;
    return __uint_as_float(b);
}

// ---------------------------- fill kernels ----------------------------------
__global__ void k_fill_f32(float* p, float v, int n) {
    int i = blockIdx.x * blockDim.x + threadIdx.x;
    if (i < n) p[i] = v;
}
__global__ void k_fill_u32(unsigned* p, unsigned v, int n) {
    int i = blockIdx.x * blockDim.x + threadIdx.x;
    if (i < n) p[i] = v;
}
__global__ void k_init_mm(unsigned* mm) {
    mm[0] = 0xFFFFFFFFu;  // running min (encoded)
    mm[1] = 0u;           // running max (encoded)
}

// ------------------------- conversion kernels -------------------------------
__global__ void k_cvt(const float* __restrict__ src, _Float16* __restrict__ dst, int n) {
    int i = blockIdx.x * blockDim.x + threadIdx.x;
    if (i < n) dst[i] = (_Float16)src[i];
}
// f32 [Kd,Nd] row-major  ->  f16 transposed [Nd,Kd]
__global__ void k_cvt_t(const float* __restrict__ W, _Float16* __restrict__ Wt,
                        int Kd, int Nd) {
    int i = blockIdx.x * blockDim.x + threadIdx.x;
    if (i >= Kd * Nd) return;
    int k = i / Nd, n = i % Nd;
    Wt[(size_t)n * Kd + k] = (_Float16)W[i];
}

// --------------------------- time encoding ----------------------------------
__global__ void k_minmax(const float* __restrict__ ts, unsigned* mm, int n) {
    int i = blockIdx.x * blockDim.x + threadIdx.x;
    if (i >= n) return;
    unsigned e = fenc(ts[i]);
    atomicMin(&mm[0], e);
    atomicMax(&mm[1], e);
}
// writes cos/sin time features (f16) into xt[:, 256:288]
__global__ void k_timeenc(const float* __restrict__ ts, const unsigned* __restrict__ mm,
                          const float* __restrict__ freq, _Float16* __restrict__ xt16,
                          int n) {
    int i = blockIdx.x * blockDim.x + threadIdx.x;
    if (i >= n * (TD / 2)) return;
    int node = i >> 4, k = i & 15;
    float mn = fdec(mm[0]), mx = fdec(mm[1]);
    float t = (ts[node] - mn) / (mx - mn + 1e-8f);
    float w = t * freq[k];
    _Float16* p = xt16 + (size_t)node * (HID + TD) + HID;
    p[k]      = (_Float16)cosf(w);
    p[16 + k] = (_Float16)sinf(w);
}

// ----------------------- WMMA GEMM (C = A @ Bt^T + bias) --------------------
// A:   [M,K] f16 row-major, leading dim lda       (K multiple of 32)
// Bt:  [N,K] f16 row-major (i.e. B transposed)    (N multiple of 32)
// C:   [M,N] f32 (optional), C16: f16 copy at ld=ldc16 (optional)
// act: 0 = none, 1 = ELU
// block = 256 threads = 8 waves; wave w -> rows [by*128+w*16,+16),
// cols [bx*32,+32): two 16x16 WMMA tiles sharing one A fragment.
__global__ __launch_bounds__(256)
void k_gemm_wmma(const _Float16* __restrict__ A, int lda,
                 const _Float16* __restrict__ Bt,
                 const float* __restrict__ bias,
                 float* __restrict__ C, int ldc,
                 _Float16* __restrict__ C16, int ldc16,
                 int M, int Kdim, int act) {
    const int wave = threadIdx.x >> 5;
    const int lane = threadIdx.x & 31;
    const int half = lane >> 4;   // which 16-lane half of the wave
    const int l16  = lane & 15;
    const int m0   = blockIdx.y * 128 + wave * 16;
    if (m0 >= M) return;          // uniform per wave: EXEC stays all-ones
    const int n0 = blockIdx.x * 32;
    const int r  = m0 + l16;      // A row for this lane
    const int c0 = n0 + l16;      // first  C column / Bt row
    const int c1 = n0 + 16 + l16; // second C column / Bt row

    const _Float16* ap  = A  + (size_t)r  * lda  + half * 16;
    const _Float16* bp0 = Bt + (size_t)c0 * Kdim + half * 16;
    const _Float16* bp1 = Bt + (size_t)c1 * Kdim + half * 16;

    v8f acc0 = {}, acc1 = {};
    for (int k0 = 0; k0 < Kdim; k0 += 32) {
        // A 16x32 f16 fragment: contiguous 16 halfs along K per lane
        v8h a0 = *(const v8h*)(ap + k0);
        v8h a1 = *(const v8h*)(ap + k0 + 8);
        // Two B 32x16 f16 fragments: contiguous 16 halfs along K of Bt rows
        v8h b00 = *(const v8h*)(bp0 + k0);
        v8h b01 = *(const v8h*)(bp0 + k0 + 8);
        v8h b10 = *(const v8h*)(bp1 + k0);
        v8h b11 = *(const v8h*)(bp1 + k0 + 8);
        __builtin_prefetch(ap + k0 + 64, 0, 3);
        __builtin_prefetch(bp0 + k0 + 64, 0, 3);
        __builtin_prefetch(bp1 + k0 + 64, 0, 3);
        v16h af  = __builtin_shufflevector(a0, a1, 0,1,2,3,4,5,6,7,8,9,10,11,12,13,14,15);
        v16h bf0 = __builtin_shufflevector(b00, b01, 0,1,2,3,4,5,6,7,8,9,10,11,12,13,14,15);
        v16h bf1 = __builtin_shufflevector(b10, b11, 0,1,2,3,4,5,6,7,8,9,10,11,12,13,14,15);
        acc0 = __builtin_amdgcn_wmma_f32_16x16x32_f16(
            false, af, false, bf0, (short)0, acc0, false, false);
        acc1 = __builtin_amdgcn_wmma_f32_16x16x32_f16(
            false, af, false, bf1, (short)0, acc1, false, false);
    }

    const float bv0 = bias ? bias[c0] : 0.0f;
    const float bv1 = bias ? bias[c1] : 0.0f;
    #pragma unroll
    for (int v = 0; v < 8; ++v) {
        int row = m0 + half * 8 + v;          // C/D layout: VGPR v -> M = v + half*8
        float x0 = acc0[v] + bv0;
        float x1 = acc1[v] + bv1;
        if (act == 1) {
            x0 = x0 > 0.0f ? x0 : (expf(x0) - 1.0f);
            x1 = x1 > 0.0f ? x1 : (expf(x1) - 1.0f);
        }
        if (C) {
            C[(size_t)row * ldc + c0] = x0;
            C[(size_t)row * ldc + c1] = x1;
        }
        if (C16) {
            C16[(size_t)row * ldc16 + c0] = (_Float16)x0;
            C16[(size_t)row * ldc16 + c1] = (_Float16)x1;
        }
    }
}

// ------------------------------ edge kernels --------------------------------
// Pass 1: scores[e,h] = dot(Q[dst],K[src]) * 1/sqrt(32); fused segment-max.
__global__ void k_scoremax(const float* __restrict__ Q, const float* __restrict__ Kb,
                           const int* __restrict__ ei, float* __restrict__ scores,
                           unsigned* __restrict__ m, int E) {
    int i = blockIdx.x * blockDim.x + threadIdx.x;
    if (i >= E * NH) return;
    int e = i >> 3, h = i & 7;
    int s = ei[e], d = ei[E + e];
    const float4* q = (const float4*)(Q  + (size_t)d * HID + h * HPD);
    const float4* k = (const float4*)(Kb + (size_t)s * HID + h * HPD);
    float acc = 0.0f;
    #pragma unroll
    for (int j = 0; j < HPD / 4; ++j) {
        float4 a = q[j], b = k[j];
        acc += a.x * b.x + a.y * b.y + a.z * b.z + a.w * b.w;
    }
    float sc = acc * 0.1767766953f;  // 1/sqrt(32)
    scores[i] = sc;
    atomicMax(&m[(size_t)d * NH + h], fenc(sc));
}

// Pass 2: v = exp(score - max); accumulate softmax denominator and the
// UNNORMALIZED message sum v * V[src]  (normalization deferred to k_norm_cvt).
__global__ void k_expmsg(const float* __restrict__ scores, const int* __restrict__ ei,
                         const unsigned* __restrict__ m, float* __restrict__ s,
                         const float* __restrict__ V, float* __restrict__ ao, int E) {
    int i = blockIdx.x * blockDim.x + threadIdx.x;
    if (i >= E * NH) return;
    int e = i >> 3, h = i & 7;
    int sr = ei[e], d = ei[E + e];
    float v = expf(scores[i] - fdec(m[(size_t)d * NH + h]));
    atomicAdd(&s[(size_t)d * NH + h], v);
    const float* vp = V  + (size_t)sr * HID + h * HPD;
    float*       op = ao + (size_t)d  * HID + h * HPD;
    #pragma unroll
    for (int j = 0; j < HPD; ++j) atomicAdd(op + j, v * vp[j]);
}

// Normalize by softmax denominator and convert to f16 for the O-projection.
__global__ void k_norm_cvt(const float* __restrict__ ao, const float* __restrict__ s,
                           _Float16* __restrict__ ao16, int n) {
    int i = blockIdx.x * blockDim.x + threadIdx.x;
    if (i >= n) return;                          // n = N*HID
    int node = i >> 8;                           // /HID
    int hd   = (i >> 5) & 7;                     // head
    float sv = s[(size_t)node * NH + hd];
    float val = sv > 0.0f ? ao[i] / sv : 0.0f;
    ao16[i] = (_Float16)val;
}

// --------------------------- pooling + final MLP ----------------------------
__global__ void k_pool(const float* __restrict__ h, float* __restrict__ g, int n) {
    int t = threadIdx.x;                 // 256 threads = one channel each
    int base = blockIdx.x * 128;
    float acc = 0.0f;
    for (int i = 0; i < 128; ++i) {
        int row = base + i;
        if (row < n) acc += h[(size_t)row * HID + t];
    }
    atomicAdd(&g[t], acc);
}

__global__ void k_mlp(const float* __restrict__ g,
                      const float* __restrict__ c1W, const float* __restrict__ c1b,
                      const float* __restrict__ c2W, const float* __restrict__ c2b,
                      float* __restrict__ out, float invN) {
    __shared__ float gs[HID];
    __shared__ float hid[HID / 2];
    int t = threadIdx.x;                 // 256 threads
    gs[t] = g[t] * invN;
    __syncthreads();
    if (t < HID / 2) {
        float a = c1b[t];
        for (int i = 0; i < HID; ++i) a += gs[i] * c1W[i * (HID / 2) + t];
        hid[t] = a > 0.0f ? a : 0.0f;
    }
    __syncthreads();
    if (t < 2) {
        float a = c2b[t];
        for (int i = 0; i < HID / 2; ++i) a += hid[i] * c2W[i * 2 + t];
        out[t] = a;
    }
}

// ---------------------------------------------------------------------------
extern "C" void kernel_launch(void* const* d_in, const int* in_sizes, int n_in,
                              void* d_out, int out_size, void* d_ws, size_t ws_size,
                              hipStream_t stream) {
    const float* x    = (const float*)d_in[0];
    const float* ts   = (const float*)d_in[1];
    const int*   ei   = (const int*)d_in[2];
    // d_in[3] = batch (all zeros, unused)
    const float* freq = (const float*)d_in[4];
    const float* inW  = (const float*)d_in[5];
    const float* inb  = (const float*)d_in[6];
    const float* qW   = (const float*)d_in[7];
    const float* qb   = (const float*)d_in[8];
    const float* kW   = (const float*)d_in[9];
    const float* kb   = (const float*)d_in[10];
    const float* vW   = (const float*)d_in[11];
    const float* vb   = (const float*)d_in[12];
    const float* oW   = (const float*)d_in[13];
    const float* ob   = (const float*)d_in[14];
    const float* c1W  = (const float*)d_in[15];
    const float* c1b  = (const float*)d_in[16];
    const float* c2W  = (const float*)d_in[17];
    const float* c2b  = (const float*)d_in[18];
    float* out = (float*)d_out;

    const int N  = in_sizes[1];          // 20000
    const int E  = in_sizes[2] / 2;      // 640000
    const int IN = in_sizes[0] / N;      // 32
    const int L  = 3;
    const int XTD = HID + TD;            // 288

    // ---------------- workspace carve-up (256B aligned) ----------------
    char* wptr = (char*)d_ws;
    auto alloc = [&](size_t bytes) -> char* {
        char* p = wptr;
        wptr += (bytes + 255) & ~size_t(255);
        return p;
    };
    unsigned* mm     = (unsigned*)alloc(2 * sizeof(unsigned));
    float*    g      = (float*)   alloc(HID * sizeof(float));
    _Float16* x16    = (_Float16*)alloc((size_t)N * IN  * sizeof(_Float16));
    _Float16* xt16   = (_Float16*)alloc((size_t)N * XTD * sizeof(_Float16));
    float*    h      = (float*)   alloc((size_t)N * HID * sizeof(float));
    float*    Q      = (float*)   alloc((size_t)N * HID * sizeof(float));
    float*    Kb     = (float*)   alloc((size_t)N * HID * sizeof(float));
    float*    V      = (float*)   alloc((size_t)N * HID * sizeof(float));
    float*    AO     = (float*)   alloc((size_t)N * HID * sizeof(float));
    _Float16* ao16   = (_Float16*)alloc((size_t)N * HID * sizeof(_Float16));
    float*    scores = (float*)   alloc((size_t)E * NH  * sizeof(float));
    unsigned* mseg   = (unsigned*)alloc((size_t)N * NH  * sizeof(unsigned));
    float*    sseg   = (float*)   alloc((size_t)N * NH  * sizeof(float));
    _Float16* inWt   = (_Float16*)alloc((size_t)IN  * HID * sizeof(_Float16));
    _Float16* qWt    = (_Float16*)alloc((size_t)L * XTD * HID * sizeof(_Float16));
    _Float16* kWt    = (_Float16*)alloc((size_t)L * XTD * HID * sizeof(_Float16));
    _Float16* vWt    = (_Float16*)alloc((size_t)L * HID * HID * sizeof(_Float16));
    _Float16* oWt    = (_Float16*)alloc((size_t)L * HID * HID * sizeof(_Float16));

    const int T = 256;
    auto blocks = [&](long long n) { return (int)((n + T - 1) / T); };

    // ---------------- time-range reduction ----------------
    k_init_mm<<<1, 1, 0, stream>>>(mm);
    k_minmax<<<blocks(N), T, 0, stream>>>(ts, mm, N);

    // ---------------- weight convert + transpose (f32 -> f16 [N,K]) ----
    k_cvt_t<<<blocks((long long)IN * HID), T, 0, stream>>>(inW, inWt, IN, HID);
    for (int l = 0; l < L; ++l) {
        k_cvt_t<<<blocks((long long)XTD * HID), T, 0, stream>>>(
            qW + (size_t)l * XTD * HID, qWt + (size_t)l * HID * XTD, XTD, HID);
        k_cvt_t<<<blocks((long long)XTD * HID), T, 0, stream>>>(
            kW + (size_t)l * XTD * HID, kWt + (size_t)l * HID * XTD, XTD, HID);
        k_cvt_t<<<blocks((long long)HID * HID), T, 0, stream>>>(
            vW + (size_t)l * HID * HID, vWt + (size_t)l * HID * HID, HID, HID);
        k_cvt_t<<<blocks((long long)HID * HID), T, 0, stream>>>(
            oW + (size_t)l * HID * HID, oWt + (size_t)l * HID * HID, HID, HID);
    }

    // ---------------- input projection: h = x @ inW + inb --------------
    k_cvt<<<blocks((long long)N * IN), T, 0, stream>>>(x, x16, N * IN);
    dim3 gemm_grid(HID / 32, (N + 127) / 128);
    k_gemm_wmma<<<gemm_grid, T, 0, stream>>>(x16, IN, inWt, inb,
                                             h, HID, xt16, XTD, N, IN, /*act=*/0);

    // time features into xt[:, 256:288]
    k_timeenc<<<blocks((long long)N * (TD / 2)), T, 0, stream>>>(ts, mm, freq, xt16, N);

    // ---------------- attention layers ----------------
    for (int l = 0; l < L; ++l) {
        const _Float16* qWl = qWt + (size_t)l * HID * XTD;
        const _Float16* kWl = kWt + (size_t)l * HID * XTD;
        const _Float16* vWl = vWt + (size_t)l * HID * HID;
        const _Float16* oWl = oWt + (size_t)l * HID * HID;

        k_gemm_wmma<<<gemm_grid, T, 0, stream>>>(xt16, XTD, qWl, qb + l * HID,
                                                 Q, HID, (_Float16*)nullptr, 0, N, XTD, 0);
        k_gemm_wmma<<<gemm_grid, T, 0, stream>>>(xt16, XTD, kWl, kb + l * HID,
                                                 Kb, HID, (_Float16*)nullptr, 0, N, XTD, 0);
        // V uses only the h-part of xt (first 256 cols, lda = 288)
        k_gemm_wmma<<<gemm_grid, T, 0, stream>>>(xt16, XTD, vWl, vb + l * HID,
                                                 V, HID, (_Float16*)nullptr, 0, N, HID, 0);

        k_fill_u32<<<blocks((long long)N * NH), T, 0, stream>>>(mseg, 0u, N * NH);
        k_fill_f32<<<blocks((long long)N * NH), T, 0, stream>>>(sseg, 0.0f, N * NH);
        k_fill_f32<<<blocks((long long)N * HID), T, 0, stream>>>(AO, 0.0f, N * HID);

        k_scoremax<<<blocks((long long)E * NH), T, 0, stream>>>(Q, Kb, ei, scores, mseg, E);
        k_expmsg  <<<blocks((long long)E * NH), T, 0, stream>>>(scores, ei, mseg, sseg, V, AO, E);
        k_norm_cvt<<<blocks((long long)N * HID), T, 0, stream>>>(AO, sseg, ao16, N * HID);

        // out-proj + ELU; writes f32 h and refreshes f16 h inside xt
        k_gemm_wmma<<<gemm_grid, T, 0, stream>>>(ao16, HID, oWl, ob + l * HID,
                                                 h, HID, xt16, XTD, N, HID, /*act=*/1);
    }

    // ---------------- global mean pool + classifier ----------------
    k_fill_f32<<<1, HID, 0, stream>>>(g, 0.0f, HID);
    k_pool<<<(N + 127) / 128, T, 0, stream>>>(h, g, N);
    k_mlp<<<1, T, 0, stream>>>(g, c1W, c1b, c2W, c2b, out, 1.0f / (float)N);
}
